// multihead_attention_40785009443477
// MI455X (gfx1250) — matrix operand order repaired and checked by reading, not compiled, not run on hardware
//
#include <hip/hip_runtime.h>
#include <hip/hip_bf16.h>

// ---------------------------------------------------------------------------
// Multi-head attention forward for MI455X (gfx1250, wave32, WMMA).
// fp32 inputs -> bf16 WMMA (v_wmma_f32_16x16x32_bf16) with fp32 accumulate.
//
// Shapes: B=2, S=2048, D=1024, H=16, Dh=64.  M = B*S = 4096.
// ---------------------------------------------------------------------------

typedef __attribute__((ext_vector_type(16))) __bf16 v16bf;
typedef __attribute__((ext_vector_type(8)))  __bf16 v8bf;
typedef __attribute__((ext_vector_type(8)))  float  v8f;
typedef __attribute__((ext_vector_type(4)))  float  v4f;

#define DIM   1024
#define SEQ   2048
#define BATCH 2
#define HEADS 16
#define HDIM  64
#define MROWS (BATCH * SEQ)   // 4096

static __device__ __forceinline__ v8f wmma_bf16(v16bf a, v16bf b, v8f c) {
  return __builtin_amdgcn_wmma_f32_16x16x32_bf16(
      /*neg_a=*/false, a, /*neg_b=*/false, b,
      /*c_mod=*/(short)0, c, /*reuse_a=*/false, /*reuse_b=*/false);
}

// 16 contiguous bf16 starting at p (p must be 16B aligned).
static __device__ __forceinline__ v16bf load16(const __bf16* p) {
  v8bf lo = *(const v8bf*)p;
  v8bf hi = *(const v8bf*)(p + 8);
  v16bf r;
#pragma unroll
  for (int i = 0; i < 8; ++i) { r[i] = lo[i]; r[i + 8] = hi[i]; }
  return r;
}

// A-matrix fragment (16x32 bf16). Lane holds row m = lane&15,
// halves j=0..15 -> K = hi*16 + j (contiguous),
// where hi = lane>>4.  `row` points at A[m][k0]; both loads are b128.
static __device__ __forceinline__ v16bf load_a_frag(const __bf16* row, int hi) {
  v8bf lo = *(const v8bf*)(row + hi * 16);
  v8bf hv = *(const v8bf*)(row + hi * 16 + 8);
  v16bf r;
#pragma unroll
  for (int i = 0; i < 8; ++i) { r[i] = lo[i]; r[i + 8] = hv[i]; }
  return r;
}

// ---------------------------------------------------------------------------
// fp32 -> bf16 elementwise convert (4 elems/thread, b64 stores)
// ---------------------------------------------------------------------------
__global__ void cvt_bf16_kernel(const float* __restrict__ in,
                                __bf16* __restrict__ out, int n4) {
  int i = blockIdx.x * blockDim.x + threadIdx.x;
  if (i >= n4) return;
  v4f x = *(const v4f*)(in + (size_t)i * 4);
  struct { __bf16 v[4]; } p;
#pragma unroll
  for (int j = 0; j < 4; ++j) p.v[j] = (__bf16)x[j];
  *(decltype(p)*)(out + (size_t)i * 4) = p;
}

// ---------------------------------------------------------------------------
// W[K,N] fp32 -> Wt[N,K] bf16 (LDS tile transpose)
// ---------------------------------------------------------------------------
__global__ void wtrans_kernel(const float* __restrict__ W,
                              __bf16* __restrict__ Wt) {
  __shared__ __bf16 tile[16][17];
  int x = blockIdx.x * 16 + threadIdx.x;  // n
  int y = blockIdx.y * 16 + threadIdx.y;  // k
  tile[threadIdx.y][threadIdx.x] = (__bf16)W[(size_t)y * DIM + x];
  __syncthreads();
  int tx = blockIdx.y * 16 + threadIdx.x; // k
  int ty = blockIdx.x * 16 + threadIdx.y; // n
  Wt[(size_t)ty * DIM + tx] = tile[threadIdx.x][threadIdx.y];
}

// ---------------------------------------------------------------------------
// GEMM: C[M=4096, N=1024] = A[M,K=1024] * Wt[N,K]^T + bias
// Block = 256 threads = 8 waves; wave w owns rows m0 = by*128 + w*16,
// cols n0 = bx*64 (4 WMMA n-tiles, A fragment reused).
// MODE 0: out bf16 [B,H,S,64]   (Q,K projections, head split)
// MODE 1: out bf16 [B,H,64,S]   (V projection, transposed for P*V operand)
// MODE 2: out fp32 [M,N]        (final O projection -> d_out)
// ---------------------------------------------------------------------------
template <int MODE>
__global__ void gemm_kernel(const __bf16* __restrict__ A,
                            const __bf16* __restrict__ Bt,
                            const float* __restrict__ bias,
                            void* __restrict__ out) {
  const int lane = threadIdx.x & 31;
  const int wave = threadIdx.x >> 5;
  const int ln = lane & 15;
  const int hi = lane >> 4;

  const int m0 = blockIdx.y * 128 + wave * 16;
  const int n0 = blockIdx.x * 64;

  v8f acc[4];
#pragma unroll
  for (int t = 0; t < 4; ++t) acc[t] = (v8f)0.0f;

  const __bf16* arow = A + (size_t)(m0 + ln) * DIM;

  for (int k0 = 0; k0 < DIM; k0 += 32) {
    v16bf af = load_a_frag(arow + k0, hi);
#pragma unroll
    for (int t = 0; t < 4; ++t) {
      // B fragment: lane holds col n = n0+16t+ln, K = hi*16 + j (contiguous)
      const __bf16* brow = Bt + (size_t)(n0 + 16 * t + ln) * DIM + k0 + hi * 16;
      v16bf bf = load16(brow);
      acc[t] = wmma_bf16(af, bf, acc[t]);
    }
  }

  // Epilogue.  C layout: lane -> col n = n0+16t+ln ; VGPR r -> row m = m0+r+8*hi
  const int mrow = m0 + 8 * hi;
  const int bI = mrow >> 11;        // / SEQ
  const int sBase = mrow & (SEQ - 1);
#pragma unroll
  for (int t = 0; t < 4; ++t) {
    const int n = n0 + 16 * t + ln;
    const float bn = bias[n];
    const int h = n >> 6;
    const int dh = n & 63;
#pragma unroll
    for (int r = 0; r < 8; ++r) {
      const float v = acc[t][r] + bn;
      const int s = sBase + r;
      if (MODE == 0) {
        __bf16* dst = (__bf16*)out;
        dst[(((size_t)bI * HEADS + h) * SEQ + s) * HDIM + dh] = (__bf16)v;
      } else if (MODE == 1) {
        __bf16* dst = (__bf16*)out;
        dst[(((size_t)bI * HEADS + h) * HDIM + dh) * SEQ + s] = (__bf16)v;
      } else {
        float* dst = (float*)out;
        dst[(size_t)(mrow + r) * DIM + n] = v;
      }
    }
  }
}

// ---------------------------------------------------------------------------
// Flash attention (transposed-score formulation).
// One wave handles one (b, h, 16-query tile) over all 2048 keys.
// S^T = K * Q^T  so the softmax row index (query) is the lane index:
// running max/sum need a single shfl_xor(16); P^T for the P*V WMMA is
// assembled register-only.  O is accumulated as O^T (4 x v8f over Dh=64).
// ---------------------------------------------------------------------------
__global__ void attention_kernel(const __bf16* __restrict__ Qh,  // [B,H,S,64]
                                 const __bf16* __restrict__ Kh,  // [B,H,S,64]
                                 const __bf16* __restrict__ Vt,  // [B,H,64,S]
                                 __bf16* __restrict__ attn) {    // [B,S,D]
  const int lane = threadIdx.x & 31;
  const int wave = threadIdx.x >> 5;
  const int ln = lane & 15;
  const int hi = lane >> 4;

  const int gw = blockIdx.x * 4 + wave;      // 0 .. 4095
  const int qtile = gw & (SEQ / 16 - 1);     // 0 .. 127
  const int bh = gw >> 7;                    // 0 .. 31
  const int h = bh & (HEADS - 1);
  const int b = bh >> 4;
  const int q0 = qtile * 16;

  const size_t headOff = ((size_t)b * HEADS + h) * SEQ;

  // Q^T B-fragments (constant over key loop): lane -> query row q0+ln,
  // fragment f covers dh = f*32 + hi*16 + {0..15} (contiguous b128 pair).
  const __bf16* qrow = Qh + (headOff + q0 + ln) * HDIM;
  v16bf qb[2];
#pragma unroll
  for (int f = 0; f < 2; ++f) qb[f] = load16(qrow + f * 32 + hi * 16);

  const __bf16* Kbase = Kh + headOff * HDIM;
  const __bf16* Vbase = Vt + ((size_t)b * HEADS + h) * HDIM * SEQ;

  float m = -1e30f;
  float l = 0.0f;
  v8f o[4];
#pragma unroll
  for (int t = 0; t < 4; ++t) o[t] = (v8f)0.0f;

  const float scale = 0.125f;  // 1/sqrt(64)

  for (int kb = 0; kb < SEQ; kb += 32) {
    // scores^T: two 16(key) x 16(query) tiles, K-dim = dh = 64 -> 2 WMMA each
    v8f st0 = (v8f)0.0f, st1 = (v8f)0.0f;
    const __bf16* krow0 = Kbase + (size_t)(kb + ln) * HDIM;
    const __bf16* krow1 = Kbase + (size_t)(kb + 16 + ln) * HDIM;
#pragma unroll
    for (int f = 0; f < 2; ++f) {
      st0 = wmma_bf16(load_a_frag(krow0 + f * 32, hi), qb[f], st0);
      st1 = wmma_bf16(load_a_frag(krow1 + f * 32, hi), qb[f], st1);
    }

    // per-lane (query = ln, half hi): keys r+8*hi (st0) and 16+r+8*hi (st1)
    float s0[8], s1[8];
    float tmax = -1e30f;
#pragma unroll
    for (int r = 0; r < 8; ++r) {
      s0[r] = st0[r] * scale;
      s1[r] = st1[r] * scale;
      tmax = fmaxf(tmax, fmaxf(s0[r], s1[r]));
    }
    tmax = fmaxf(tmax, __shfl_xor(tmax, 16));

    const float mnew = fmaxf(m, tmax);
    const float corr = __expf(m - mnew);
    float p0[8], p1[8];
    float rsum = 0.0f;
#pragma unroll
    for (int r = 0; r < 8; ++r) {
      p0[r] = __expf(s0[r] - mnew);
      p1[r] = __expf(s1[r] - mnew);
      rsum += p0[r] + p1[r];
    }
    rsum += __shfl_xor(rsum, 16);
    l = l * corr + rsum;
    m = mnew;
#pragma unroll
    for (int t = 0; t < 4; ++t)
#pragma unroll
      for (int r = 0; r < 8; ++r) o[t][r] *= corr;

    // Build P^T B-fragment (32 keys x 16 queries), bf16.
    // B layout: lane n = query, halves j -> key = hi*16 + j.
    // hi=0 lane: j<8 own st0, j>=8 partner st0 ; hi=1: j<8 partner st1, j>=8 own st1.
    v16bf pb;
#pragma unroll
    for (int r = 0; r < 8; ++r) {
      const float o0 = __shfl_xor(p0[r], 16);
      const float o1 = __shfl_xor(p1[r], 16);
      const float vlo = hi ? o1 : p0[r];
      const float vhi = hi ? p1[r] : o0;
      pb[r] = (__bf16)vlo;
      pb[r + 8] = (__bf16)vhi;
    }

    // O^T += V^T * P^T : A = V^T tile (16 dh x 32 keys), contiguous rows in Vt
#pragma unroll
    for (int t = 0; t < 4; ++t) {
      const __bf16* vrow = Vbase + (size_t)(16 * t + ln) * SEQ + kb;
      o[t] = wmma_bf16(load_a_frag(vrow, hi), pb, o[t]);
    }
  }

  // Normalize and store: attn[b][q][h*64 + dh], dh = 16t + 8*hi + r (packed b128)
  const float inv = 1.0f / l;
  __bf16* obase = attn + ((size_t)b * SEQ + q0 + ln) * DIM + h * HDIM;
#pragma unroll
  for (int t = 0; t < 4; ++t) {
    v8bf pack;
#pragma unroll
    for (int r = 0; r < 8; ++r) pack[r] = (__bf16)(o[t][r] * inv);
    *(v8bf*)(obase + 16 * t + 8 * hi) = pack;
  }
}

// ---------------------------------------------------------------------------
// Host launch
// ---------------------------------------------------------------------------
extern "C" void kernel_launch(void* const* d_in, const int* in_sizes, int n_in,
                              void* d_out, int out_size, void* d_ws, size_t ws_size,
                              hipStream_t stream) {
  (void)in_sizes; (void)n_in; (void)out_size; (void)ws_size;

  const float* q_in = (const float*)d_in[0];
  const float* k_in = (const float*)d_in[1];
  const float* v_in = (const float*)d_in[2];
  const float* Wq = (const float*)d_in[3];
  const float* bq = (const float*)d_in[4];
  const float* Wk = (const float*)d_in[5];
  const float* bk = (const float*)d_in[6];
  const float* Wv = (const float*)d_in[7];
  const float* bv = (const float*)d_in[8];
  const float* Wo = (const float*)d_in[9];
  const float* bo = (const float*)d_in[10];

  const size_t ACT = (size_t)MROWS * DIM;          // 4194304 elems
  const size_t WEL = (size_t)DIM * DIM;            // 1048576 elems
  char* ws = (char*)d_ws;
  __bf16* Xq   = (__bf16*)(ws);
  __bf16* Xk   = (__bf16*)(ws + 8u  * 1024 * 1024);
  __bf16* Xv   = (__bf16*)(ws + 16u * 1024 * 1024);
  __bf16* Wqt  = (__bf16*)(ws + 24u * 1024 * 1024);
  __bf16* Wkt  = (__bf16*)(ws + 26u * 1024 * 1024);
  __bf16* Wvt  = (__bf16*)(ws + 28u * 1024 * 1024);
  __bf16* Wot  = (__bf16*)(ws + 30u * 1024 * 1024);
  __bf16* Qh   = (__bf16*)(ws + 32u * 1024 * 1024);
  __bf16* Kh   = (__bf16*)(ws + 40u * 1024 * 1024);
  __bf16* Vt   = (__bf16*)(ws + 48u * 1024 * 1024);
  __bf16* attn = (__bf16*)(ws + 56u * 1024 * 1024);

  // 1) convert activations to bf16
  {
    int n4 = (int)(ACT / 4);
    dim3 g((n4 + 255) / 256), blk(256);
    cvt_bf16_kernel<<<g, blk, 0, stream>>>(q_in, Xq, n4);
    cvt_bf16_kernel<<<g, blk, 0, stream>>>(k_in, Xk, n4);
    cvt_bf16_kernel<<<g, blk, 0, stream>>>(v_in, Xv, n4);
    (void)WEL;
  }
  // 2) transpose+convert weights
  {
    dim3 g(DIM / 16, DIM / 16), blk(16, 16);
    wtrans_kernel<<<g, blk, 0, stream>>>(Wq, Wqt);
    wtrans_kernel<<<g, blk, 0, stream>>>(Wk, Wkt);
    wtrans_kernel<<<g, blk, 0, stream>>>(Wv, Wvt);
    wtrans_kernel<<<g, blk, 0, stream>>>(Wo, Wot);
  }
  // 3) Q/K/V projections (head-split; V transposed)
  {
    dim3 g(DIM / 64, MROWS / 128), blk(256);
    gemm_kernel<0><<<g, blk, 0, stream>>>(Xq, Wqt, bq, (void*)Qh);
    gemm_kernel<0><<<g, blk, 0, stream>>>(Xk, Wkt, bk, (void*)Kh);
    gemm_kernel<1><<<g, blk, 0, stream>>>(Xv, Wvt, bv, (void*)Vt);
  }
  // 4) flash attention -> attn [B,S,D] bf16
  {
    dim3 g((BATCH * HEADS * (SEQ / 16)) / 4), blk(128);
    attention_kernel<<<g, blk, 0, stream>>>(Qh, Kh, Vt, attn);
  }
  // 5) output projection -> fp32 d_out
  {
    dim3 g(DIM / 64, MROWS / 128), blk(256);
    gemm_kernel<2><<<g, blk, 0, stream>>>(attn, Wot, bo, d_out);
  }
}
